// Decoder_lfa_4217657885150
// MI455X (gfx1250) — compile-verified, matrix operand order checked by reading, not run on hardware
//
#include <hip/hip_runtime.h>
#include <hip/hip_bf16.h>
#include <stdint.h>

typedef __attribute__((ext_vector_type(16))) __bf16 v16bf;
typedef __attribute__((ext_vector_type(8)))  __bf16 v8bf;
typedef __attribute__((ext_vector_type(8)))  float  v8f;

#define B_    2
#define Q_    16384
#define KNB   16
#define ND    4096
#define NS    16384
#define DLB   128
#define DSC   64
#define DMID  74
#define DATT  138
#define DNL   128
#define KT    5          // 5 * 32 = 160 padded K
#define NT    9          // 9 * 16 = 144 padded N
#define KPAD  160
#define NPAD  144
#define EPSV  1e-5f

// ---------- bf16 helpers (bit-exact RNE, no reliance on __bf16 arithmetic) ----------
__device__ __forceinline__ __bf16 f2bf(float f) {
    unsigned u = __builtin_bit_cast(unsigned, f);
    unsigned r = u + 0x7FFFu + ((u >> 16) & 1u);
    unsigned short h = (unsigned short)(r >> 16);
    return __builtin_bit_cast(__bf16, h);
}
__device__ __forceinline__ float bf2f(__bf16 b) {
    unsigned short h = __builtin_bit_cast(unsigned short, b);
    unsigned u = ((unsigned)h) << 16;
    return __builtin_bit_cast(float, u);
}

// LDS byte offset of a __shared__ object (flat shared pointer: low 32 bits = LDS offset)
__device__ __forceinline__ unsigned lds_off(const void* p) {
    return (unsigned)(uintptr_t)p;
}

// Issue one wave-wide async copy: 32 lanes x 16B = 512B global -> LDS (ASYNCcnt tracked)
__device__ __forceinline__ void async_copy_512(unsigned lds_addr, uint64_t gaddr) {
    asm volatile("global_load_async_to_lds_b128 %0, %1, off"
                 :: "v"(lds_addr), "v"(gaddr)
                 : "memory");
}
__device__ __forceinline__ void wait_async0() {
    asm volatile("s_wait_asynccnt 0" ::: "memory");
}

// ---------- weight fragment prep: W (o,c) -> B-matrix tiles (K=c, N=o), bf16 ----------
//  lanes 0-15 : N = lane,     slots 0..7 -> K 0..7,  slots 8..15 -> K 16..23
//  lanes 16-31: N = lane-16,  slots 0..7 -> K 8..15, slots 8..15 -> K 24..31
__global__ void prep_wfrag(const float* __restrict__ Wfc1,
                           const float* __restrict__ Wfc2,
                           __bf16* __restrict__ frag1,
                           __bf16* __restrict__ frag2) {
    int tid = blockIdx.x * blockDim.x + threadIdx.x;
    const int per = KT * NT * 32;
    if (tid >= 2 * per) return;
    const float* W = (tid < per) ? Wfc1 : Wfc2;
    __bf16* F      = (tid < per) ? frag1 : frag2;
    int t    = tid % per;
    int lane = t & 31;
    int tile = t >> 5;
    int kt = tile / NT, nt = tile % NT;
    int nloc = lane & 15;
    int koff = (lane & 16) ? 8 : 0;
    __bf16* dst = F + ((size_t)tile * 32 + lane) * 16;
#pragma unroll
    for (int s = 0; s < 16; s++) {
        int kl = (s < 8) ? (koff + s) : (koff + 8 + s);   // s>=8 -> koff+16+(s-8)
        int k = kt * 32 + kl;
        int n = nt * 16 + nloc;
        float v = (k < DATT && n < DATT) ? W[n * DATT + k] : 0.0f;
        dst[s] = f2bf(v);
    }
}

// ---------- A fragment from LDS X[m][KPAD] (16-bit A layout, two aligned b128 loads) ----------
__device__ __forceinline__ v16bf load_afrag(const __bf16* __restrict__ X,
                                            int mbase, int kt, int lane) {
    int m    = mbase + (lane & 15);
    int koff = kt * 32 + ((lane & 16) ? 8 : 0);
    const __bf16* p = X + m * KPAD + koff;
    v8bf lo = *(const v8bf*)(p);        // K offsets +0..7
    v8bf hi = *(const v8bf*)(p + 16);   // K offsets +16..23
    v16bf a;
#pragma unroll
    for (int i = 0; i < 8; i++) { a[i] = lo[i]; a[8 + i] = hi[i]; }
    return a;
}

// ======================= Stage 1: f_hat -> att_pool1 -> f_tilde =======================
__global__ void __launch_bounds__(32)
stage1_kernel(const float* __restrict__ feats_lb, const float* __restrict__ feats_sc,
              const int* __restrict__ NM_lb, const int* __restrict__ NM_sc,
              const float* __restrict__ coords_lb, const float* __restrict__ coords_sc,
              const float* __restrict__ coords_q,
              const float* __restrict__ W_lb, const float* __restrict__ g_lb,
              const float* __restrict__ b_lb, const float* __restrict__ m_lb,
              const float* __restrict__ v_lb,
              const float* __restrict__ W_sc, const float* __restrict__ g_sc,
              const float* __restrict__ b_sc, const float* __restrict__ m_sc,
              const float* __restrict__ v_sc,
              const __bf16* __restrict__ wfrag1, const float* __restrict__ W_mlp1,
              const float* __restrict__ g1, const float* __restrict__ b1,
              const float* __restrict__ m1, const float* __restrict__ v1,
              float* __restrict__ ftilde) {
    __shared__ __align__(16) __bf16 Xl[32][KPAD];       // f_hat row-block, bf16
    __shared__ __align__(16) float  st_lb[KNB * DLB];   // async staging (8 KB)
    __shared__ __align__(16) float  st_sc[KNB * DSC];   // async staging (4 KB)
    __shared__ float attl[32][NPAD];                    // attention logits
    __shared__ float aggl[DATT];                        // pooled vector

    const int lane = threadIdx.x;
    const int bq   = blockIdx.x;
    const int b    = bq / Q_;

    // ---- kick off async global->LDS streaming of this query's feature block ----
    {
        uint64_t g_lb = (uint64_t)(uintptr_t)(feats_lb + (size_t)bq * KNB * DLB)
                        + (unsigned)(lane * 16);
        unsigned l_lb = lds_off(&st_lb[0]) + (unsigned)(lane * 16);
#pragma unroll
        for (int r = 0; r < 16; r++)                     // 16 x 512B = 8 KB
            async_copy_512(l_lb + r * 512, g_lb + (uint64_t)(r * 512));
        uint64_t g_sc = (uint64_t)(uintptr_t)(feats_sc + (size_t)bq * KNB * DSC)
                        + (unsigned)(lane * 16);
        unsigned l_sc = lds_off(&st_sc[0]) + (unsigned)(lane * 16);
#pragma unroll
        for (int r = 0; r < 8; r++)                      // 8 x 512B = 4 KB
            async_copy_512(l_sc + r * 512, g_sc + (uint64_t)(r * 512));
    }

    const float qx = coords_q[(size_t)bq * 3 + 0];
    const float qy = coords_q[(size_t)bq * 3 + 1];
    const float qz = coords_q[(size_t)bq * 3 + 2];

    // ---- rel-pos encoding + tiny matmul + BN + ReLU, overlapped with async copy ----
    {
        const int  j     = lane & 15;
        const bool is_sc = lane >= 16;
        int idx = is_sc ? NM_sc[(size_t)bq * KNB + j] : NM_lb[(size_t)bq * KNB + j];
        const float* cs = is_sc ? (coords_sc + ((size_t)b * NS + idx) * 3)
                                : (coords_lb + ((size_t)b * ND + idx) * 3);
        float nx = cs[0], ny = cs[1], nz = cs[2];
        float rx = qx - nx, ry = qy - ny, rz = qz - nz;
        float dis = sqrtf(rx * rx + ry * ry + rz * rz);
        float enc[10] = {dis, rx, ry, rz, qx, qy, qz, nx, ny, nz};
        if (!is_sc) {
            for (int o = 0; o < 10; o++) {
                float y = 0.f;
#pragma unroll
                for (int c = 0; c < 10; c++) y += enc[c] * W_lb[o * 10 + c];
                float s = g_lb[o] * rsqrtf(v_lb[o] + EPSV);
                y = (y - m_lb[o]) * s + b_lb[o];
                Xl[j][o] = f2bf(fmaxf(y, 0.f));
            }
        } else {
            for (int o = 0; o < DMID; o++) {
                float y = 0.f;
#pragma unroll
                for (int c = 0; c < 10; c++) y += enc[c] * W_sc[o * 10 + c];
                float s = g_sc[o] * rsqrtf(v_sc[o] + EPSV);
                y = (y - m_sc[o]) * s + b_sc[o];
                Xl[16 + j][o] = f2bf(fmaxf(y, 0.f));
            }
        }
        for (int c = DATT; c < KPAD; c++) Xl[lane][c] = f2bf(0.f);  // K padding
    }

    // ---- drain async copies, convert staged f32 features to bf16 rows of X ----
    wait_async0();
    for (int r = 0; r < KNB; r++) {
        float4 v = *(const float4*)(st_lb + r * DLB + lane * 4);
        int c0 = 10 + lane * 4;
        Xl[r][c0 + 0] = f2bf(v.x); Xl[r][c0 + 1] = f2bf(v.y);
        Xl[r][c0 + 2] = f2bf(v.z); Xl[r][c0 + 3] = f2bf(v.w);
    }
    for (int r = 0; r < KNB; r++) {
        float2 v = *(const float2*)(st_sc + r * DSC + lane * 2);
        int c0 = DMID + lane * 2;
        Xl[16 + r][c0 + 0] = f2bf(v.x); Xl[16 + r][c0 + 1] = f2bf(v.y);
    }
    __syncthreads();

    // ---- att = X(32x160) @ Wfc1^T via v_wmma_f32_16x16x32_bf16 ----
    // nt outer, kt inner, both M-tiles accumulated per B fragment:
    // each W tile is loaded once and feeds two back-to-back WMMAs; the ten
    // A fragments are nt-invariant and get hoisted into VGPRs by LICM.
    for (int nt = 0; nt < NT; nt++) {
        v8f acc0 = {}, acc1 = {};
#pragma unroll
        for (int kt = 0; kt < KT; kt++) {
            v16bf bb = *(const v16bf*)(wfrag1 + (((size_t)kt * NT + nt) * 32 + lane) * 16);
            v16bf a0 = load_afrag(&Xl[0][0],  0, kt, lane);
            v16bf a1 = load_afrag(&Xl[0][0], 16, kt, lane);
            acc0 = __builtin_amdgcn_wmma_f32_16x16x32_bf16(
                     false, a0, false, bb, (short)0, acc0, false, false);
            acc1 = __builtin_amdgcn_wmma_f32_16x16x32_bf16(
                     false, a1, false, bb, (short)0, acc1, false, false);
        }
        int n0 = nt * 16 + (lane & 15);
        int mb = (lane & 16) ? 8 : 0;
#pragma unroll
        for (int r = 0; r < 8; r++) {
            attl[mb + r][n0]      = acc0[r];
            attl[16 + mb + r][n0] = acc1[r];
        }
    }
    __syncthreads();

    // ---- per-channel softmax over 32 neighbors + weighted aggregation ----
    for (int c = lane; c < DATT; c += 32) {
        float mx = -3.4e38f;
        for (int n = 0; n < 32; n++) mx = fmaxf(mx, attl[n][c]);
        float s = 0.f, w = 0.f;
        for (int n = 0; n < 32; n++) {
            float e = __expf(attl[n][c] - mx);
            s += e;
            w += bf2f(Xl[n][c]) * e;
        }
        aggl[c] = w / s;
    }
    __syncthreads();

    // ---- MLP1 (128x138) + BN + ReLU -> f_tilde ----
    float* outp = ftilde + (size_t)bq * DNL;
    for (int o = lane; o < DNL; o += 32) {
        float y = 0.f;
        for (int c = 0; c < DATT; c++) y += aggl[c] * W_mlp1[o * DATT + c];
        float s = g1[o] * rsqrtf(v1[o] + EPSV);
        y = (y - m1[o]) * s + b1[o];
        outp[o] = fmaxf(y, 0.f);
    }
}

// ======================= Stage 2: cluster round -> output =======================
__global__ void __launch_bounds__(32)
stage2_kernel(const int* __restrict__ NM_cl, const float* __restrict__ coords_q,
              const float* __restrict__ W_cl, const float* __restrict__ g_cl,
              const float* __restrict__ b_cl, const float* __restrict__ m_cl,
              const float* __restrict__ v_cl,
              const __bf16* __restrict__ wfrag2, const float* __restrict__ W_mlp2,
              const float* __restrict__ g2, const float* __restrict__ b2,
              const float* __restrict__ m2, const float* __restrict__ v2,
              const float* __restrict__ ftilde, float* __restrict__ out) {
    __shared__ __align__(16) __bf16 Xl[16][KPAD];
    __shared__ __align__(16) float  st_f[KNB * DNL];    // async gather staging (8 KB)
    __shared__ float attl[16][NPAD];
    __shared__ float aggl[DATT];

    const int lane = threadIdx.x;
    const int bq   = blockIdx.x;
    const int b    = bq / Q_;
    const int q    = bq % Q_;

    const int* nm = NM_cl + (size_t)bq * KNB;

    // ---- async gather: each neighbor row of f_tilde is one contiguous 512B copy ----
    {
        unsigned l_f = lds_off(&st_f[0]) + (unsigned)(lane * 16);
        for (int r = 0; r < KNB; r++) {
            int idx = nm[r];
            uint64_t g = (uint64_t)(uintptr_t)(ftilde + ((size_t)b * Q_ + idx) * DNL)
                         + (unsigned)(lane * 16);
            async_copy_512(l_f + r * 512, g);
        }
    }

    const float qx = coords_q[(size_t)bq * 3 + 0];
    const float qy = coords_q[(size_t)bq * 3 + 1];
    const float qz = coords_q[(size_t)bq * 3 + 2];

    if (lane < KNB) {
        int idx = nm[lane];
        const float* cs = coords_q + ((size_t)b * Q_ + idx) * 3;
        float nx = cs[0], ny = cs[1], nz = cs[2];
        float rx = qx - nx, ry = qy - ny, rz = qz - nz;
        float dis = sqrtf(rx * rx + ry * ry + rz * rz);
        float enc[10] = {dis, rx, ry, rz, qx, qy, qz, nx, ny, nz};
        for (int o = 0; o < 10; o++) {
            float y = 0.f;
#pragma unroll
            for (int c = 0; c < 10; c++) y += enc[c] * W_cl[o * 10 + c];
            float s = g_cl[o] * rsqrtf(v_cl[o] + EPSV);
            y = (y - m_cl[o]) * s + b_cl[o];
            Xl[lane][o] = f2bf(fmaxf(y, 0.f));
        }
        for (int c = DATT; c < KPAD; c++) Xl[lane][c] = f2bf(0.f);
    }

    // ---- drain async gather, convert to bf16 rows ----
    wait_async0();
    for (int r = 0; r < KNB; r++) {
        float4 v = *(const float4*)(st_f + r * DNL + lane * 4);
        int c0 = 10 + lane * 4;
        Xl[r][c0 + 0] = f2bf(v.x); Xl[r][c0 + 1] = f2bf(v.y);
        Xl[r][c0 + 2] = f2bf(v.z); Xl[r][c0 + 3] = f2bf(v.w);
    }
    __syncthreads();

    // att2 = X2(16x160) @ Wfc2^T
    for (int nt = 0; nt < NT; nt++) {
        v8f acc = {};
#pragma unroll
        for (int kt = 0; kt < KT; kt++) {
            v16bf a  = load_afrag(&Xl[0][0], 0, kt, lane);
            v16bf bb = *(const v16bf*)(wfrag2 + (((size_t)kt * NT + nt) * 32 + lane) * 16);
            acc = __builtin_amdgcn_wmma_f32_16x16x32_bf16(
                    false, a, false, bb, (short)0, acc, false, false);
        }
        int n0 = nt * 16 + (lane & 15);
        int mb = (lane & 16) ? 8 : 0;
#pragma unroll
        for (int r = 0; r < 8; r++) attl[mb + r][n0] = acc[r];
    }
    __syncthreads();

    for (int c = lane; c < DATT; c += 32) {
        float mx = -3.4e38f;
        for (int n = 0; n < KNB; n++) mx = fmaxf(mx, attl[n][c]);
        float s = 0.f, w = 0.f;
        for (int n = 0; n < KNB; n++) {
            float e = __expf(attl[n][c] - mx);
            s += e;
            w += bf2f(Xl[n][c]) * e;
        }
        aggl[c] = w / s;
    }
    __syncthreads();

    // MLP2 + BN + ReLU, store transposed (B,128,Q,1)
    for (int o = lane; o < DNL; o += 32) {
        float y = 0.f;
        for (int c = 0; c < DATT; c++) y += aggl[c] * W_mlp2[o * DATT + c];
        float s = g2[o] * rsqrtf(v2[o] + EPSV);
        y = (y - m2[o]) * s + b2[o];
        out[((size_t)b * DNL + o) * Q_ + q] = fmaxf(y, 0.f);
    }
}

// ======================= host launcher =======================
extern "C" void kernel_launch(void* const* d_in, const int* in_sizes, int n_in,
                              void* d_out, int out_size, void* d_ws, size_t ws_size,
                              hipStream_t stream) {
    (void)in_sizes; (void)n_in; (void)out_size; (void)ws_size;
    const float* feats_lb = (const float*)d_in[0];
    const float* feats_sc = (const float*)d_in[1];
    const int*   NM_lb    = (const int*)d_in[2];
    const int*   NM_sc    = (const int*)d_in[3];
    const float* c_lb     = (const float*)d_in[4];
    const float* c_sc     = (const float*)d_in[5];
    const float* c_q      = (const float*)d_in[6];
    const int*   NM_cl    = (const int*)d_in[7];
    const float* W_lb     = (const float*)d_in[8];
    const float* g_lb     = (const float*)d_in[9];
    const float* b_lb     = (const float*)d_in[10];
    const float* m_lb     = (const float*)d_in[11];
    const float* v_lb     = (const float*)d_in[12];
    const float* W_sc     = (const float*)d_in[13];
    const float* g_sc     = (const float*)d_in[14];
    const float* b_sc     = (const float*)d_in[15];
    const float* m_sc     = (const float*)d_in[16];
    const float* v_sc     = (const float*)d_in[17];
    const float* W_fc1    = (const float*)d_in[18];
    const float* W_mlp1   = (const float*)d_in[19];
    const float* g1       = (const float*)d_in[20];
    const float* b1       = (const float*)d_in[21];
    const float* m1       = (const float*)d_in[22];
    const float* v1       = (const float*)d_in[23];
    const float* W_cl     = (const float*)d_in[24];
    const float* g_cl     = (const float*)d_in[25];
    const float* b_cl     = (const float*)d_in[26];
    const float* m_cl     = (const float*)d_in[27];
    const float* v_cl     = (const float*)d_in[28];
    const float* W_fc2    = (const float*)d_in[29];
    const float* W_mlp2   = (const float*)d_in[30];
    const float* g2       = (const float*)d_in[31];
    const float* b2       = (const float*)d_in[32];
    const float* m2       = (const float*)d_in[33];
    const float* v2       = (const float*)d_in[34];

    char* ws = (char*)d_ws;
    const size_t fragBytes = (size_t)KT * NT * 32 * 16 * sizeof(unsigned short); // 46080
    __bf16* frag1  = (__bf16*)(ws);
    __bf16* frag2  = (__bf16*)(ws + fragBytes);
    float*  ftilde = (float*)(ws + 2 * fragBytes);   // B*Q*128 floats = 16 MB

    const int tot = 2 * KT * NT * 32;                // 2880 prep threads
    prep_wfrag<<<(tot + 127) / 128, 128, 0, stream>>>(W_fc1, W_fc2, frag1, frag2);

    stage1_kernel<<<B_ * Q_, 32, 0, stream>>>(
        feats_lb, feats_sc, NM_lb, NM_sc, c_lb, c_sc, c_q,
        W_lb, g_lb, b_lb, m_lb, v_lb,
        W_sc, g_sc, b_sc, m_sc, v_sc,
        frag1, W_mlp1, g1, b1, m1, v1, ftilde);

    stage2_kernel<<<B_ * Q_, 32, 0, stream>>>(
        NM_cl, c_q, W_cl, g_cl, b_cl, m_cl, v_cl,
        frag2, W_mlp2, g2, b2, m2, v2, ftilde, (float*)d_out);
}
